// AttributeDecoupledGNN_32031866093971
// MI455X (gfx1250) — compile-verified, matrix operand order checked by reading, not run on hardware
//
#include <hip/hip_runtime.h>
#include <math.h>

#define HDIM 128

typedef __attribute__((ext_vector_type(16))) _Float16 v16h;
typedef __attribute__((ext_vector_type(8)))  _Float16 v8h;
typedef __attribute__((ext_vector_type(2)))  _Float16 v2h;
typedef __attribute__((ext_vector_type(8)))  float    v8f;

__device__ __forceinline__ v8f wmma_16x16x32_f16(v16h a, v16h b, v8f c) {
    // D = A(16x32 f16) * B(32x16 f16) + C(16x16 f32)
    return __builtin_amdgcn_wmma_f32_16x16x32_f16(
        /*neg_a=*/false, a, /*neg_b=*/false, b,
        /*c_mod=*/(short)0, c, /*reuse_a=*/false, /*reuse_b=*/false);
}

// A-tile fragment (16x32 f16 layout, ISA 7.12.2): lane half selects kbase 0/8;
// halves [0..7] = K kbase..+7, [8..15] = K kbase+16..+23. Four b128 global loads.
__device__ __forceinline__ v16h load_a_frag(const float* __restrict__ ap) {
    float4 u0 = *(const float4*)(ap + 0);
    float4 u1 = *(const float4*)(ap + 4);
    float4 u2 = *(const float4*)(ap + 16);
    float4 u3 = *(const float4*)(ap + 20);
    v16h a;
    a[0]=(_Float16)u0.x;  a[1]=(_Float16)u0.y;  a[2]=(_Float16)u0.z;  a[3]=(_Float16)u0.w;
    a[4]=(_Float16)u1.x;  a[5]=(_Float16)u1.y;  a[6]=(_Float16)u1.z;  a[7]=(_Float16)u1.w;
    a[8]=(_Float16)u2.x;  a[9]=(_Float16)u2.y;  a[10]=(_Float16)u2.z; a[11]=(_Float16)u2.w;
    a[12]=(_Float16)u3.x; a[13]=(_Float16)u3.y; a[14]=(_Float16)u3.z; a[15]=(_Float16)u3.w;
    return a;
}

// Stage W (row-major [KD][HDIM] f32 in global) into LDS as f16, column-major
// with XOR-swizzled 16B chunks: half-addr = col*KD + ((k>>3 ^ (col&15))<<3) + (k&7).
// Conflict-free ds_store_b32 (paired halves) and later conflict-free ds_load_b128.
template<int KD>
__device__ __forceinline__ void stage_weights(_Float16* __restrict__ lw,
                                              const float* __restrict__ W, int tid) {
    constexpr int TOT = (KD / 2) * HDIM;
    for (int i = tid; i < TOT; i += 256) {
        const int col = i & (HDIM - 1);        // consecutive tids -> consecutive cols (coalesced)
        const int k   = (i >> 7) << 1;         // even k
        v2h p;
        p[0] = (_Float16)W[k * HDIM + col];
        p[1] = (_Float16)W[(k + 1) * HDIM + col];
        const int addr = col * KD + ((((k >> 3) ^ (col & 15)) << 3) | (k & 7));
        *(v2h*)(lw + addr) = p;
    }
}

// B-tile fragment (32x16 f16): lane's column fixed; two swizzled ds_load_b128.
template<int KD>
__device__ __forceinline__ v16h load_b_frag(const _Float16* __restrict__ lw,
                                            int k0, int kbase, int col) {
    const int swz = col & 15;
    const int c0 = ((k0 + kbase) >> 3) ^ swz;
    const int c1 = ((k0 + kbase + 16) >> 3) ^ swz;
    const _Float16* base = lw + col * KD;
    v8h lo = *(const v8h*)(base + (c0 << 3));
    v8h hi = *(const v8h*)(base + (c1 << 3));
    return __builtin_shufflevector(lo, hi, 0,1,2,3,4,5,6,7,8,9,10,11,12,13,14,15);
}

// out[n,128] = act(A1[n,KD] @ W1[KD,128] (+ A2[n,128] @ W2[128,128]) + bias)
// Block: 256 threads = 8 waves, owns 128 rows x 128 cols.
// Wave w owns rows [blockBase+16w, +16); sweeps all 8 column tiles, so one
// A-fragment feeds 8 WMMAs and every A row is loaded by exactly one wave.
template<int KD, bool RELU, bool DUAL>
__global__ __launch_bounds__(256, 1)
void gemm_wmma_kernel(const float* __restrict__ A1, const float* __restrict__ W1,
                      const float* __restrict__ A2, const float* __restrict__ W2,
                      const float* __restrict__ bias, float* __restrict__ out, int n)
{
    constexpr int W1E = KD * HDIM;
    constexpr int LDSE = W1E + (DUAL ? 128 * HDIM : 0);
    __shared__ _Float16 lW[LDSE];

    const int tid = threadIdx.x;
    const int lane  = tid & 31;
    const int wave  = tid >> 5;
    const int l16   = lane & 15;
    const int half  = lane >> 4;        // 0: lanes 0-15, 1: lanes 16-31
    const int kbase = half << 3;

    const long long blockBase = (long long)blockIdx.x << 7;     // 128 rows per block
    const long long waveRow0  = blockBase + ((long long)wave << 4);
    long long aRow = waveRow0 + l16;
    if (aRow >= n) aRow = n - 1;        // clamp loads; EXEC stays all-1s for WMMA

    const float* a1p = A1 + aRow * KD;
    const float* a2p = DUAL ? (A2 + aRow * 128) : A1;

    // pull the A stream toward the WGP while we stage weights in LDS
    __builtin_prefetch(a1p, 0, 0);
    if (DUAL) __builtin_prefetch(a2p, 0, 0);

    stage_weights<KD>(lW, W1, tid);
    if (DUAL) stage_weights<128>(lW + W1E, W2, tid);
    __syncthreads();

    v8f acc[8];
#pragma unroll
    for (int ct = 0; ct < 8; ++ct) acc[ct] = (v8f){};

#pragma unroll
    for (int k0 = 0; k0 < KD; k0 += 32) {
        {
            v16h a = load_a_frag(a1p + k0 + kbase);
#pragma unroll
            for (int ct = 0; ct < 8; ++ct) {
                v16h b = load_b_frag<KD>(lW, k0, kbase, (ct << 4) + l16);
                acc[ct] = wmma_16x16x32_f16(a, b, acc[ct]);
            }
        }
        if (DUAL) {
            v16h a2 = load_a_frag(a2p + k0 + kbase);
#pragma unroll
            for (int ct = 0; ct < 8; ++ct) {
                v16h b2 = load_b_frag<128>(lW + W1E, k0, kbase, (ct << 4) + l16);
                acc[ct] = wmma_16x16x32_f16(a2, b2, acc[ct]);
            }
        }
    }

#pragma unroll
    for (int ct = 0; ct < 8; ++ct) {
        const int col = (ct << 4) + l16;
        const float bv = bias[col];
#pragma unroll
        for (int v = 0; v < 8; ++v) {
            long long m = waveRow0 + v + (half << 3);  // C/D layout: VGPR v -> row v (+8 upper)
            if (m < n) {
                float val = acc[ct][v] + bv;
                if (RELU) val = fmaxf(val, 0.0f);
                out[m * HDIM + col] = val;
            }
        }
    }
}

__global__ void deg_kernel(const int* __restrict__ ei, long long E, float* __restrict__ deg) {
    long long i = (long long)blockIdx.x * blockDim.x + threadIdx.x;
    if (i < E) atomicAdd(&deg[ei[E + i]], 1.0f);
}

// 32 threads per edge, each moves 4 contiguous floats: gather h[src], atomic-add to s[dst]
__global__ void scatter_kernel(const int* __restrict__ ei, long long E,
                               const float* __restrict__ h, float* __restrict__ s) {
    long long idx = (long long)blockIdx.x * blockDim.x + threadIdx.x;
    if (idx >= E * 32) return;
    long long e = idx >> 5;
    int c = (int)(idx & 31) << 2;
    long long src = ei[e];
    long long dst = ei[E + e];
    float4 v = *(const float4*)(h + src * HDIM + c);
    float* p = s + dst * HDIM + c;
    atomicAdd(p + 0, v.x); atomicAdd(p + 1, v.y);
    atomicAdd(p + 2, v.z); atomicAdd(p + 3, v.w);
}

__global__ void divide_kernel(float* __restrict__ s, const float* __restrict__ deg, int total) {
    int idx = blockIdx.x * blockDim.x + threadIdx.x;
    if (idx >= total) return;
    s[idx] = s[idx] / fmaxf(deg[idx >> 7], 1.0f);
}

// y = relu(e[n,5] @ W0[5,128] + b0) — K=5, pure VALU
__global__ void edge_mlp0_kernel(const float* __restrict__ e, const float* __restrict__ W0,
                                 const float* __restrict__ b0, float* __restrict__ y, int total) {
    int idx = blockIdx.x * blockDim.x + threadIdx.x;
    if (idx >= total) return;
    int r = idx >> 7, c = idx & 127;
    float acc = b0[c];
#pragma unroll
    for (int k = 0; k < 5; ++k) acc = fmaf(e[r * 5 + k], W0[k * HDIM + c], acc);
    y[idx] = fmaxf(acc, 0.0f);
}

// out[n] = sigmoid(concat(feat,dist) @ final_W + final_b)
__global__ void final_kernel(const float* __restrict__ feat, const float* __restrict__ dist,
                             const float* __restrict__ fW, const float* __restrict__ fb,
                             float* __restrict__ out, int n) {
    int r = blockIdx.x * blockDim.x + threadIdx.x;
    if (r >= n) return;
    float acc = fb[0];
    const float4* fp = (const float4*)(feat + (long long)r * HDIM);
    const float4* dp = (const float4*)(dist + (long long)r * HDIM);
    const float4* w0 = (const float4*)fW;
    const float4* w1 = (const float4*)(fW + HDIM);
#pragma unroll 4
    for (int i = 0; i < HDIM / 4; ++i) {
        float4 a = fp[i], w = w0[i];
        acc = fmaf(a.x, w.x, acc); acc = fmaf(a.y, w.y, acc);
        acc = fmaf(a.z, w.z, acc); acc = fmaf(a.w, w.w, acc);
    }
#pragma unroll 4
    for (int i = 0; i < HDIM / 4; ++i) {
        float4 a = dp[i], w = w1[i];
        acc = fmaf(a.x, w.x, acc); acc = fmaf(a.y, w.y, acc);
        acc = fmaf(a.z, w.z, acc); acc = fmaf(a.w, w.w, acc);
    }
    out[r] = 1.0f / (1.0f + expf(-acc));
}

extern "C" void kernel_launch(void* const* d_in, const int* in_sizes, int n_in,
                              void* d_out, int out_size, void* d_ws, size_t ws_size,
                              hipStream_t stream) {
    const float* x     = (const float*)d_in[0];
    const int*   ei    = (const int*)d_in[1];
    const float* eattr = (const float*)d_in[2];
    const float* pre_W = (const float*)d_in[3];
    const float* pre_b = (const float*)d_in[4];
    const float* c1_Ws = (const float*)d_in[5];
    const float* c1_Wn = (const float*)d_in[6];
    const float* c1_b  = (const float*)d_in[7];
    const float* c2_Ws = (const float*)d_in[8];
    const float* c2_Wn = (const float*)d_in[9];
    const float* c2_b  = (const float*)d_in[10];
    const float* np_W  = (const float*)d_in[11];
    const float* np_b  = (const float*)d_in[12];
    const float* dW0   = (const float*)d_in[13];
    const float* db0   = (const float*)d_in[14];
    const float* dW1   = (const float*)d_in[15];
    const float* db1   = (const float*)d_in[16];
    const float* dW2   = (const float*)d_in[17];
    const float* db2   = (const float*)d_in[18];
    const float* dW3   = (const float*)d_in[19];
    const float* db3   = (const float*)d_in[20];
    const float* finW  = (const float*)d_in[21];
    const float* finB  = (const float*)d_in[22];
    float* out = (float*)d_out;

    const int n = in_sizes[0] / 256;               // F_IN = 256
    const long long E = (long long)in_sizes[1] / 2;

    float* bufA = (float*)d_ws;
    float* bufB = bufA + (size_t)n * HDIM;
    float* bufC = bufB + (size_t)n * HDIM;
    float* deg  = bufC + (size_t)n * HDIM;

    const unsigned rowBlocks = (unsigned)((n + 127) / 128);   // 128 rows per block
    const unsigned elemBlocks = (unsigned)(((long long)n * HDIM + 255) / 256);
    const unsigned edgeBlocks = (unsigned)((E + 255) / 256);
    const unsigned scatBlocks = (unsigned)((E * 32 + 255) / 256);
    const size_t nhBytes = (size_t)n * HDIM * sizeof(float);
    dim3 blk(256);

    // degrees (graph is static across both aggregation rounds)
    hipMemsetAsync(deg, 0, (size_t)n * sizeof(float), stream);
    deg_kernel<<<edgeBlocks, blk, 0, stream>>>(ei, E, deg);

    // h = x @ pre_W + pre_b                         -> bufA
    gemm_wmma_kernel<256, false, false><<<rowBlocks, blk, 0, stream>>>(
        x, pre_W, nullptr, nullptr, pre_b, bufA, n);

    // agg = mean_agg(h)                             -> bufB
    hipMemsetAsync(bufB, 0, nhBytes, stream);
    scatter_kernel<<<scatBlocks, blk, 0, stream>>>(ei, E, bufA, bufB);
    divide_kernel<<<elemBlocks, blk, 0, stream>>>(bufB, deg, n * HDIM);

    // h1 = relu(h@c1_Ws + agg@c1_Wn + c1_b)         -> bufC   (fused dual GEMM)
    gemm_wmma_kernel<128, true, true><<<rowBlocks, blk, 0, stream>>>(
        bufA, c1_Ws, bufB, c1_Wn, c1_b, bufC, n);

    // agg = mean_agg(h1)                            -> bufB
    hipMemsetAsync(bufB, 0, nhBytes, stream);
    scatter_kernel<<<scatBlocks, blk, 0, stream>>>(ei, E, bufC, bufB);
    divide_kernel<<<elemBlocks, blk, 0, stream>>>(bufB, deg, n * HDIM);

    // h2 = relu(h1@c2_Ws + agg@c2_Wn + c2_b)        -> bufA
    gemm_wmma_kernel<128, true, true><<<rowBlocks, blk, 0, stream>>>(
        bufC, c2_Ws, bufB, c2_Wn, c2_b, bufA, n);

    // feat = h2 @ nodepost_W + nodepost_b           -> bufC
    gemm_wmma_kernel<128, false, false><<<rowBlocks, blk, 0, stream>>>(
        bufA, np_W, nullptr, nullptr, np_b, bufC, n);

    // dist path: y0 -> bufA, y1 -> bufB, y2 -> bufA, dist -> bufB
    edge_mlp0_kernel<<<elemBlocks, blk, 0, stream>>>(eattr, dW0, db0, bufA, n * HDIM);
    gemm_wmma_kernel<128, true, false><<<rowBlocks, blk, 0, stream>>>(
        bufA, dW1, nullptr, nullptr, db1, bufB, n);
    gemm_wmma_kernel<128, true, false><<<rowBlocks, blk, 0, stream>>>(
        bufB, dW2, nullptr, nullptr, db2, bufA, n);
    gemm_wmma_kernel<128, false, false><<<rowBlocks, blk, 0, stream>>>(
        bufA, dW3, nullptr, nullptr, db3, bufB, n);

    // out = sigmoid(concat(feat, dist) @ final_W + final_b)
    final_kernel<<<(unsigned)((n + 255) / 256), blk, 0, stream>>>(
        bufC, bufB, finW, finB, out, n);
}
